// VQVAE_47914655154527
// MI455X (gfx1250) — matrix operand order repaired and checked by reading, not compiled, not run on hardware
//
// VQ-VAE forward for MI455X (gfx1250): all heavy math via v_wmma_f32_16x16x32_bf16
// (f32 accumulate). Intermediates + weights + codebook in bf16 (halves HBM
// traffic). Convs/deconvs = implicit GEMM, double-buffered LDS staging with
// register-held gathers so global loads overlap WMMA; VQ = WMMA GEMM vs bf16
// codebook with software-pipelined B fragments + cross-lane argmin reduction.
//
// Workspace layout (~172 MB):
//   bufA  : 134217728 B  bf16 h1 (32x128x128x128), later dec-h2
//   bufB  :  33554432 B  bf16 h2 (32x128x64x64),   later dec-h1
//   idx   :    524288 B  131072 int32 VQ indices
//   cbn   :      2048 B  512 codebook squared norms (f32)
//   cbbf  :    524288 B  bf16 codebook
//   w*bf  :  ~3.5 MB     bf16 weights
#include <hip/hip_runtime.h>
#include <cstdint>

typedef __attribute__((ext_vector_type(16))) __bf16 v16bf;
typedef __attribute__((ext_vector_type(2)))  __bf16 v2bf;
typedef __attribute__((ext_vector_type(8)))  float  v8f;

union FragU { v16bf v; uint4 q[2]; uint32_t u[8]; };

static __device__ inline uint32_t pack_bf2(float lo, float hi) {
  v2bf t;
  t[0] = (__bf16)lo;
  t[1] = (__bf16)hi;
  return __builtin_bit_cast(uint32_t, t);   // v_cvt_pk_bf16_f32
}
static __device__ inline uint16_t bfbits_f(float f) {
  __bf16 h = (__bf16)f;
  return __builtin_bit_cast(uint16_t, h);
}
static __device__ inline uint16_t ld_bf(const float* p)    { return bfbits_f(*p); }
static __device__ inline uint16_t ld_bf(const uint16_t* p) { return *p; }
static __device__ inline void st_out(float* p, float v)    { *p = v; }
static __device__ inline void st_out(uint16_t* p, float v) { *p = bfbits_f(v); }

// A fragment (16x32 bf16, MxK): lanes 0-15 M=0..15, khalf=lane>>4; VGPR r holds
// pair k = 16*khalf + 2*r. One contiguous 32B span per lane.
static __device__ inline v16bf load_frag_a(const __bf16* base, int m, int khalf,
                                           int rowStrideHalf, int k0) {
  FragU f;
  const char* p = (const char*)(base + (size_t)m * rowStrideHalf + k0);
  f.q[0] = *(const uint4*)(p + khalf * 32);
  f.q[1] = *(const uint4*)(p + khalf * 32 + 16);
  return f.v;
}

// B fragment (32x16 bf16, KxN) from LDS stored [n][k] (row stride 32 halfwords).
static __device__ inline v16bf load_frag_b_lds(const __bf16* base, int n, int khalf) {
  FragU f;
  const char* p = (const char*)(base + (size_t)n * 32 + khalf * 16);
  f.q[0] = *(const uint4*)p;
  f.q[1] = *(const uint4*)(p + 16);
  return f.v;
}

// B fragment column from a global bf16 vector: two b128 loads, zero VALU.
static __device__ inline v16bf load_frag_b_glbf(const uint16_t* col, int khalf) {
  FragU f;
  const uint4* s = (const uint4*)(col + khalf * 16);
  f.q[0] = s[0];
  f.q[1] = s[1];
  return f.v;
}

static __device__ inline v8f wmma_bf16(v16bf a, v16bf b, v8f c) {
  return __builtin_amdgcn_wmma_f32_16x16x32_bf16(false, a, false, b,
                                                 (short)0, c, false, false);
}

// Elementwise f32 -> bf16 copy (weights/codebook prep). n multiple of 4.
__global__ __launch_bounds__(256) void f2bf_copy(const float* __restrict__ s,
                                                 uint16_t* __restrict__ d, int n4) {
  int t = blockIdx.x * 256 + threadIdx.x;
  if (t < n4) {
    float4 v = ((const float4*)s)[t];
    uint2 o;
    o.x = pack_bf2(v.x, v.y);
    o.y = pack_bf2(v.z, v.w);
    ((uint2*)d)[t] = o;
  }
}

// ---------------------------------------------------------------------------
// Implicit-GEMM conv, NCHW activations, OIHW weights (pre-converted bf16).
// Block = 8 waves along M (tile 128M x 64N): 4 WMMAs/wave/K-step share one A
// fragment. Double-buffered LDS; panel s+1 gathered into registers while the
// WMMAs for panel s execute. Requires OW % 64 == 0, KREAL % 16 == 0.
// ---------------------------------------------------------------------------
template <typename TIN, typename TOUT, int CIN, int KH, int KW, int STRIDE,
          int PAD, bool RELU>
__global__ __launch_bounds__(256) void conv_wmma(
    const TIN* __restrict__ x, const uint16_t* __restrict__ wbf,
    const float* __restrict__ bias, TOUT* __restrict__ y,
    int NB, int H, int W, int COUT, int OH, int OW) {
  constexpr int KREAL = CIN * KH * KW;
  constexpr int KPAD  = (KREAL + 31) & ~31;
  constexpr int STEPS = KPAD / 32;
  static_assert(KREAL % 16 == 0, "A staging assumes 16-element K chunks");
  __shared__ __align__(16) __bf16 sA[2][128 * 32];  // [m][k]
  __shared__ __align__(16) __bf16 sB[2][64 * 32];   // [n][k]

  const int tid = threadIdx.x, wave = tid >> 5, lane = tid & 31;
  const int khalf = lane >> 4, nl = lane & 15;
  const int m0 = blockIdx.y * 128;
  const int n0 = blockIdx.x * 64;
  const int nb  = n0 / (OH * OW);
  const int rem = n0 % (OH * OW);
  const int oh  = rem / OW;
  const int ow0 = rem % OW;
  const int mlA = tid >> 1, halfA = tid & 1;        // A-staging slot

  auto gatherA = [&](int k0, uint4& v0, uint4& v1) {
    int m = m0 + mlA, k = k0 + halfA * 16;
    v0 = make_uint4(0u, 0u, 0u, 0u); v1 = v0;
    if (m < COUT && k < KREAL) {
      const uint4* s = (const uint4*)(wbf + (size_t)m * KREAL + k);
      v0 = s[0]; v1 = s[1];
    }
  };
  auto gatherB = [&](int k0, uint16_t bv[8]) {
#pragma unroll
    for (int i = 0; i < 8; ++i) {
      int e = tid + i * 256;
      int nn = e & 63, kl = e >> 6;
      int k = k0 + kl;
      uint16_t v = 0;
      if (k < KREAL) {
        int ci = k / (KH * KW);
        int rr = k % (KH * KW);
        int kh = rr / KW, kw = rr % KW;
        int ih = oh * STRIDE - PAD + kh;
        int iw = (ow0 + nn) * STRIDE - PAD + kw;
        if (ih >= 0 && ih < H && iw >= 0 && iw < W)
          v = ld_bf(x + (((size_t)nb * CIN + ci) * H + ih) * W + iw);
      }
      bv[i] = v;
    }
  };

  uint4 av0, av1;
  uint16_t bv[8];
  gatherA(0, av0, av1);
  gatherB(0, bv);

  v8f acc[4] = {{}, {}, {}, {}};
  for (int s = 0; s < STEPS; ++s) {
    __bf16* cA = sA[s & 1];
    __bf16* cB = sB[s & 1];
    {
      uint4* d = (uint4*)((char*)cA + mlA * 64 + halfA * 32);
      d[0] = av0; d[1] = av1;
    }
#pragma unroll
    for (int i = 0; i < 8; ++i) {
      int e = tid + i * 256;
      ((uint16_t*)cB)[(e & 63) * 32 + (e >> 6)] = bv[i];
    }
    __syncthreads();
    if (s + 1 < STEPS) {                 // prefetch next panel into registers
      gatherA((s + 1) * 32, av0, av1);
      gatherB((s + 1) * 32, bv);
    }
    v16bf a = load_frag_a(cA, wave * 16 + nl, khalf, 32, 0);
#pragma unroll
    for (int t = 0; t < 4; ++t) {
      v16bf b = load_frag_b_lds(cB + (size_t)t * 16 * 32, nl, khalf);
      acc[t] = wmma_bf16(a, b, acc[t]);
    }
    __syncthreads();
  }

#pragma unroll
  for (int t = 0; t < 4; ++t) {
    int ow = ow0 + t * 16 + nl;
#pragma unroll
    for (int r = 0; r < 8; ++r) {
      int m = m0 + wave * 16 + r + 8 * khalf;
      if (m < COUT) {
        float v = acc[t][r] + bias[m];
        if (RELU) v = fmaxf(v, 0.f);
        st_out(y + (((size_t)nb * COUT + m) * OH + oh) * OW + ow, v);
      }
    }
  }
}

// ---------------------------------------------------------------------------
// Transposed conv (k=4, stride 2, pad 1 equivalent) as 4 parity-class stride-1
// 2x2 convs. Weights IOHW (pre-converted bf16). gridDim.z = parity (ph*2+pw).
// Block = 8 waves along N (tile 32M x 128N); double-buffered LDS pipeline.
// ---------------------------------------------------------------------------
template <typename TIN, typename TOUT, int CIN, int COUT, int IH, int IW, bool RELU>
__global__ __launch_bounds__(256) void deconv_wmma(
    const TIN* __restrict__ x, const uint16_t* __restrict__ wbf,
    const float* __restrict__ bias, TOUT* __restrict__ y) {
  constexpr int K = CIN * 4;          // 2x2 taps per input channel
  constexpr int STEPS = K / 32;
  __shared__ __align__(16) __bf16 sA[2][32 * 32];    // [m][k]
  __shared__ __align__(16) __bf16 sB[2][128 * 32];   // [n][k]

  const int tid = threadIdx.x, wave = tid >> 5, lane = tid & 31;
  const int khalf = lane >> 4, nl = lane & 15;
  const int ph = (int)blockIdx.z >> 1, pw = (int)blockIdx.z & 1;
  const int m0 = blockIdx.y * 32;
  const int n0 = blockIdx.x * 128;

  auto gatherA = [&](int k0, uint16_t av[4]) {
#pragma unroll
    for (int i = 0; i < 4; ++i) {
      int e = tid + i * 256;
      int ml = e >> 5, kl = e & 31;
      int k = k0 + kl, m = m0 + ml;
      int ci = k >> 2, t4 = k & 3, dih = t4 >> 1, diw = t4 & 1;
      int wh = ph ? (2 - 2 * dih) : (3 - 2 * dih);
      int ww = pw ? (2 - 2 * diw) : (3 - 2 * diw);
      uint16_t v = 0;
      if (m < COUT) v = wbf[(((size_t)ci * COUT + m) * 4 + wh) * 4 + ww];
      av[i] = v;
    }
  };
  auto gatherB = [&](int k0, uint16_t bv[16]) {
#pragma unroll
    for (int i = 0; i < 16; ++i) {
      int e = tid + i * 256;
      int nn = e & 127, kl = e >> 7;
      int k = k0 + kl;
      int ci = k >> 2, t4 = k & 3, dih = t4 >> 1, diw = t4 & 1;
      int n = n0 + nn;
      int nb = n / (IH * IW), rem = n % (IH * IW);
      int a_ = rem / IW, b_ = rem % IW;
      int ih = a_ + (ph ? dih : dih - 1);
      int iw = b_ + (pw ? diw : diw - 1);
      uint16_t v = 0;
      if (ih >= 0 && ih < IH && iw >= 0 && iw < IW)
        v = ld_bf(x + (((size_t)nb * CIN + ci) * IH + ih) * IW + iw);
      bv[i] = v;
    }
  };

  uint16_t av[4], bv[16];
  gatherA(0, av);
  gatherB(0, bv);

  v8f acc0 = {}, acc1 = {};
  for (int s = 0; s < STEPS; ++s) {
    __bf16* cA = sA[s & 1];
    __bf16* cB = sB[s & 1];
#pragma unroll
    for (int i = 0; i < 4; ++i) {
      int e = tid + i * 256;
      ((uint16_t*)cA)[(e >> 5) * 32 + (e & 31)] = av[i];
    }
#pragma unroll
    for (int i = 0; i < 16; ++i) {
      int e = tid + i * 256;
      ((uint16_t*)cB)[(e & 127) * 32 + (e >> 7)] = bv[i];
    }
    __syncthreads();
    if (s + 1 < STEPS) {                 // prefetch next panel into registers
      gatherA((s + 1) * 32, av);
      gatherB((s + 1) * 32, bv);
    }
    v16bf b  = load_frag_b_lds(cB + (size_t)wave * 16 * 32, nl, khalf);
    v16bf a0 = load_frag_a(cA, nl, khalf, 32, 0);
    v16bf a1 = load_frag_a(cA, 16 + nl, khalf, 32, 0);
    acc0 = wmma_bf16(a0, b, acc0);
    acc1 = wmma_bf16(a1, b, acc1);
    __syncthreads();
  }

  const int n = n0 + wave * 16 + nl;
  const int nb = n / (IH * IW), rem = n % (IH * IW);
  const int a_ = rem / IW, b_ = rem % IW;
  const int oh = 2 * a_ + ph, ow = 2 * b_ + pw;
#pragma unroll
  for (int half = 0; half < 2; ++half) {
    const v8f& acc = half ? acc1 : acc0;
#pragma unroll
    for (int r = 0; r < 8; ++r) {
      int m = m0 + half * 16 + r + 8 * khalf;
      if (m < COUT) {
        float v = acc[r] + bias[m];
        if (RELU) v = fmaxf(v, 0.f);
        st_out(y + (((size_t)nb * COUT + m) * (2 * IH) + oh) * (2 * IW) + ow, v);
      }
    }
  }
}

// ---------------------------------------------------------------------------
// VQ: squared norms of codebook rows (f32 codebook).
// ---------------------------------------------------------------------------
__global__ void cb_norm_k(const float* __restrict__ cb, float* __restrict__ cbn) {
  int c = blockIdx.x * blockDim.x + threadIdx.x;
  if (c < 512) {
    float s = 0.f;
    const float* p = cb + (size_t)c * 512;
    for (int d = 0; d < 512; ++d) s += p[d] * p[d];
    cbn[c] = s;
  }
}

// ---------------------------------------------------------------------------
// VQ argmin: each wave owns 16 flat rows (contiguous 512-f32 chunks of z_e),
// stages them as bf16 in LDS once (A fragments stay VGPR-resident), then WMMA
// vs all 512 codes with software-pipelined B fragments from the bf16 codebook.
// Tracks min of ||c||^2 - 2*dot per row; cross-lane min+index reduction.
// Block = 128 threads (4 waves), 64 KB LDS.
// ---------------------------------------------------------------------------
__global__ __launch_bounds__(128) void vq_argmin(
    const float* __restrict__ z, const uint16_t* __restrict__ cbbf,
    const float* __restrict__ cbn, int* __restrict__ idx) {
  __shared__ __align__(16) __bf16 sZ[4 * 16 * 512];
  const int wave = threadIdx.x >> 5, lane = threadIdx.x & 31;
  const int khalf = lane >> 4, nl = lane & 15;
  const int j0 = blockIdx.x * 64 + wave * 16;
  __bf16* myZ = sZ + (size_t)wave * 16 * 512;
  uint32_t* myZ32 = (uint32_t*)myZ;

  for (int i = 0; i < 128; ++i) {
    int p  = i * 32 + lane;        // pair index, 256 pairs per row
    int m  = p >> 8;
    int kp = p & 255;
    const float2 src = *(const float2*)(z + (size_t)(j0 + m) * 512 + kp * 2);
    myZ32[m * 256 + kp] = pack_bf2(src.x, src.y);
  }
  __syncthreads();

  float bestv[8];
  int   besti[8];
#pragma unroll
  for (int r = 0; r < 8; ++r) { bestv[r] = 3.4e38f; besti[r] = 0; }

  for (int nt = 0; nt < 32; ++nt) {
    v8f acc = {};
    const int c = nt * 16 + nl;
    const uint16_t* cbcol = cbbf + (size_t)c * 512;
    v16bf bcur = load_frag_b_glbf(cbcol, khalf);
#pragma unroll
    for (int ks = 0; ks < 16; ++ks) {
      v16bf bnext = bcur;
      if (ks < 15) bnext = load_frag_b_glbf(cbcol + (ks + 1) * 32, khalf);
      v16bf a = load_frag_a(myZ, nl, khalf, 512, ks * 32);
      acc = wmma_bf16(a, bcur, acc);
      bcur = bnext;
    }
    const float cn = cbn[c];
#pragma unroll
    for (int r = 0; r < 8; ++r) {
      float s = cn - 2.f * acc[r];
      if (s < bestv[r]) { bestv[r] = s; besti[r] = c; }
    }
  }

  for (int off = 1; off < 16; off <<= 1) {
#pragma unroll
    for (int r = 0; r < 8; ++r) {
      float ov = __shfl_xor(bestv[r], off);
      int   oi = __shfl_xor(besti[r], off);
      if (ov < bestv[r] || (ov == bestv[r] && oi < besti[r])) {
        bestv[r] = ov; besti[r] = oi;
      }
    }
  }
  if (nl == 0) {
#pragma unroll
    for (int r = 0; r < 8; ++r) idx[j0 + r + 8 * khalf] = besti[r];
  }
}

// z_q[j][:] = codebook[idx[j]][:]   (float4 copies from f32 codebook)
__global__ __launch_bounds__(256) void zq_scatter(
    const float* __restrict__ cb, const int* __restrict__ idx,
    float* __restrict__ zq) {
  size_t t = (size_t)blockIdx.x * 256 + threadIdx.x;
  int row = (int)(t >> 7);
  int d4  = (int)(t & 127);
  const float4* src = (const float4*)(cb + (size_t)idx[row] * 512);
  ((float4*)(zq + (size_t)row * 512))[d4] = src[d4];
}

// ---------------------------------------------------------------------------
extern "C" void kernel_launch(void* const* d_in, const int* in_sizes, int n_in,
                              void* d_out, int out_size, void* d_ws, size_t ws_size,
                              hipStream_t stream) {
  (void)in_sizes; (void)n_in; (void)out_size; (void)ws_size;
  const float* x      = (const float*)d_in[0];
  const float* enc_w1 = (const float*)d_in[1];
  const float* enc_b1 = (const float*)d_in[2];
  const float* enc_w2 = (const float*)d_in[3];
  const float* enc_b2 = (const float*)d_in[4];
  const float* enc_w3 = (const float*)d_in[5];
  const float* enc_b3 = (const float*)d_in[6];
  const float* cbk    = (const float*)d_in[7];
  const float* dec_w1 = (const float*)d_in[8];
  const float* dec_b1 = (const float*)d_in[9];
  const float* dec_w2 = (const float*)d_in[10];
  const float* dec_b2 = (const float*)d_in[11];
  const float* dec_w3 = (const float*)d_in[12];
  const float* dec_b3 = (const float*)d_in[13];

  // Outputs: (x_recon, z_e, z_q) concatenated.
  float* x_recon = (float*)d_out;
  float* z_e = x_recon + (size_t)32 * 3 * 256 * 256;       // +6291456
  float* z_q = z_e + (size_t)32 * 512 * 64 * 64;           // +67108864

  // Workspace (bf16 intermediates).
  char* ws = (char*)d_ws;
  uint16_t* bufA = (uint16_t*)ws;                              // 134217728 B
  uint16_t* bufB = (uint16_t*)(ws + 134217728u);               //  33554432 B
  int*      idxb = (int*)     (ws + 134217728u + 33554432u);   //    524288 B
  float*    cbn  = (float*)   (ws + 167772160u + 524288u);     //      2048 B
  uint16_t* cbbf = (uint16_t*)(ws + 168296448u + 2048u);       //    524288 B
  char*     wbase = ws + 168298496u + 524288u;
  uint16_t* w1bf  = (uint16_t*)(wbase);                        //  12288 B
  uint16_t* w2bf  = (uint16_t*)(wbase + 12288u);               // 524288 B
  uint16_t* w3bf  = (uint16_t*)(wbase + 536576u);              // 1179648 B
  uint16_t* dw1bf = (uint16_t*)(wbase + 1716224u);             // 1179648 B
  uint16_t* dw2bf = (uint16_t*)(wbase + 2895872u);             // 524288 B
  uint16_t* dw3bf = (uint16_t*)(wbase + 3420160u);             //  12288 B

  dim3 blk(256);

  // Prep: bf16 weights + codebook, codebook norms.
  f2bf_copy<<<dim3((6144 / 4 + 255) / 256), blk, 0, stream>>>(enc_w1, w1bf, 6144 / 4);
  f2bf_copy<<<dim3((262144 / 4 + 255) / 256), blk, 0, stream>>>(enc_w2, w2bf, 262144 / 4);
  f2bf_copy<<<dim3((589824 / 4 + 255) / 256), blk, 0, stream>>>(enc_w3, w3bf, 589824 / 4);
  f2bf_copy<<<dim3((589824 / 4 + 255) / 256), blk, 0, stream>>>(dec_w1, dw1bf, 589824 / 4);
  f2bf_copy<<<dim3((262144 / 4 + 255) / 256), blk, 0, stream>>>(dec_w2, dw2bf, 262144 / 4);
  f2bf_copy<<<dim3((6144 / 4 + 255) / 256), blk, 0, stream>>>(dec_w3, dw3bf, 6144 / 4);
  f2bf_copy<<<dim3((262144 / 4 + 255) / 256), blk, 0, stream>>>(cbk, cbbf, 262144 / 4);
  cb_norm_k<<<2, 256, 0, stream>>>(cbk, cbn);

  // Encoder.
  conv_wmma<float, uint16_t, 3, 4, 4, 2, 1, true>
      <<<dim3(8192, 1, 1), blk, 0, stream>>>(
          x, w1bf, enc_b1, bufA, 32, 256, 256, 128, 128, 128);
  conv_wmma<uint16_t, uint16_t, 128, 4, 4, 2, 1, true>
      <<<dim3(2048, 1, 1), blk, 0, stream>>>(
          bufA, w2bf, enc_b2, bufB, 32, 128, 128, 128, 64, 64);
  conv_wmma<uint16_t, float, 128, 3, 3, 1, 1, false>
      <<<dim3(2048, 4, 1), blk, 0, stream>>>(
          bufB, w3bf, enc_b3, z_e, 32, 64, 64, 512, 64, 64);

  // Vector quantize.
  vq_argmin<<<2048, 128, 0, stream>>>(z_e, cbbf, cbn, idxb);
  zq_scatter<<<65536, 256, 0, stream>>>(cbk, idxb, z_q);

  // Decoder.
  conv_wmma<float, uint16_t, 512, 3, 3, 1, 1, true>
      <<<dim3(2048, 1, 1), blk, 0, stream>>>(
          z_q, dw1bf, dec_b1, bufB, 32, 64, 64, 128, 64, 64);
  deconv_wmma<uint16_t, uint16_t, 128, 128, 64, 64, true>
      <<<dim3(1024, 4, 4), blk, 0, stream>>>(bufB, dw2bf, dec_b2, bufA);
  deconv_wmma<uint16_t, float, 128, 3, 128, 128, false>
      <<<dim3(4096, 1, 4), blk, 0, stream>>>(bufA, dw3bf, dec_b3, x_recon);
}
